// AdditiveAttention_11596411699905
// MI455X (gfx1250) — matrix operand order repaired and checked by reading, not compiled, not run on hardware
//
#include <hip/hip_runtime.h>
#include <hip/hip_bf16.h>

#define Bn 32
#define Vn 4096
#define Hn 1024
#define ROWS 64           // key rows per workgroup: 134KB LDS -> 2 WGs co-resident per WGP
#define MT   (ROWS / 16)  // M-tiles per workgroup
#define PITCH 1032        // LDS row pitch in bf16 elements (2064 B: 16B-aligned, bank-spread)

typedef __attribute__((ext_vector_type(16))) __bf16 v16bf;
typedef __attribute__((ext_vector_type(8)))  float  v8f;

struct __align__(16) U4 { unsigned int x, y, z, w; };
struct U4x2 { U4 lo, hi; };

__device__ __forceinline__ unsigned short f32_to_bf16(float f) {
  unsigned int u = __builtin_bit_cast(unsigned int, f);
  u += 0x7FFFu + ((u >> 16) & 1u);          // round-to-nearest-even
  return (unsigned short)(u >> 16);
}

// ---------------- prep: qp[b,o] = bias[o] + query[b,:] . Wq[o,:] ----------------
__global__ __launch_bounds__(256) void qp_kernel(const float* __restrict__ query,
                                                 const float* __restrict__ Wq,
                                                 const float* __restrict__ bias,
                                                 float* __restrict__ qp) {
  __shared__ float q[Hn];
  const int b = blockIdx.x, tid = threadIdx.x;
  for (int h = tid; h < Hn; h += 256) q[h] = query[b * Hn + h];
  __syncthreads();
  for (int o = tid; o < Hn; o += 256) {
    const float* w = Wq + (size_t)o * Hn;
    float acc = bias[o];
    for (int h = 0; h < Hn; h += 4) {
      float4 wv = *(const float4*)(w + h);
      acc += q[h] * wv.x + q[h + 1] * wv.y + q[h + 2] * wv.z + q[h + 3] * wv.w;
    }
    qp[b * Hn + o] = acc;
  }
}

// ---------------- prep: Wk (f32, row-major over h) -> bf16 ----------------
__global__ __launch_bounds__(256) void wkb_kernel(const float* __restrict__ Wk,
                                                  unsigned short* __restrict__ wkb) {
  const int i = (blockIdx.x * 256 + threadIdx.x) * 4;
  float4 f = *(const float4*)(Wk + i);
  unsigned int p0 = (unsigned int)f32_to_bf16(f.x) | ((unsigned int)f32_to_bf16(f.y) << 16);
  unsigned int p1 = (unsigned int)f32_to_bf16(f.z) | ((unsigned int)f32_to_bf16(f.w) << 16);
  *(uint2*)(wkb + i) = make_uint2(p0, p1);
}

// ---------------- fused score kernel ----------------
// score[b,v] = sum_o Ws[o] * tanh( key[b,v,:] . Wk[o,:] + qp[b,o] )   (bs dropped: softmax-invariant)
__global__ __launch_bounds__(256) void score_kernel(const float* __restrict__ key,
                                                    const unsigned short* __restrict__ wkb,
                                                    const float* __restrict__ qp,
                                                    const float* __restrict__ Wsv,
                                                    float* __restrict__ score) {
  extern __shared__ unsigned char smem[];
  unsigned short* sKey = (unsigned short*)smem;                    // ROWS x PITCH bf16
  float* sPart = (float*)(smem + ROWS * PITCH * 2);                // 8 waves x ROWS partials

  const int tilesPerB = Vn / ROWS;
  const int b    = blockIdx.x / tilesPerB;
  const int v0   = (blockIdx.x % tilesPerB) * ROWS;
  const int tid  = threadIdx.x;
  const int lane = tid & 31;
  const int wave = tid >> 5;
  const int col  = lane & 15;                  // C-matrix column / A,B lane role
  const int hi   = lane >> 4;                  // 0: lanes 0-15, 1: lanes 16-31

  for (int i = tid; i < 8 * ROWS; i += 256) sPart[i] = 0.0f;

  // stage ROWS key rows (f32 -> bf16) into LDS, coalesced float4 reads
  const float* kbase = key + ((size_t)b * Vn + v0) * Hn;
  for (int i = tid; i < ROWS * (Hn / 4); i += 256) {
    const int e = i << 2;
    const int r = e >> 10;                     // / Hn
    const int c = e & (Hn - 1);
    float4 f = *(const float4*)(kbase + e);
    unsigned int p0 = (unsigned int)f32_to_bf16(f.x) | ((unsigned int)f32_to_bf16(f.y) << 16);
    unsigned int p1 = (unsigned int)f32_to_bf16(f.z) | ((unsigned int)f32_to_bf16(f.w) << 16);
    *(uint2*)&sKey[r * PITCH + c] = make_uint2(p0, p1);
  }
  __syncthreads();

  const v8f vzero = {0.f, 0.f, 0.f, 0.f, 0.f, 0.f, 0.f, 0.f};

  // each wave owns 8 N-tiles (128 output columns); loops all MT M-tiles
  for (int nt = 0; nt < 8; ++nt) {
    const int o0 = (wave * 8 + nt) * 16;
    const float wsw = Wsv[o0 + col];
    const float qpw = qp[b * Hn + o0 + col];

    v8f C[MT];
#pragma unroll
    for (int m = 0; m < MT; ++m) C[m] = vzero;

    // B fragment: lane holds column o0+col; K = k..k+15 (lanes<16) or k+16..k+31.
    // Software-pipelined one k-step ahead so global-load latency hides under WMMAs.
    const unsigned short* wkBase = wkb + (size_t)(o0 + col) * Hn + (hi << 4);
    U4x2 bfrag;
    bfrag.lo = *(const U4*)(wkBase);
    bfrag.hi = *(const U4*)(wkBase + 8);

    for (int k = 0; k < Hn; k += 32) {
      const int kn = (k + 32) & (Hn - 1);      // wraps on last iter (harmless reload, no branch)
      U4x2 bnext;
      bnext.lo = *(const U4*)(wkBase + kn);
      bnext.hi = *(const U4*)(wkBase + kn + 8);

      const v16bf Bm = __builtin_bit_cast(v16bf, bfrag);

      // Preload ALL MT A-fragments for this k-step (one clause of ds_load_b128,
      // single wait), then issue the MT independent WMMAs back-to-back.
      // A fragment: lane holds row m*16+col; K = k..k+15 (lanes<16)
      //                                      K = k+16..k+31 (lanes>=16)
      const int aoff = k + (hi << 4);
      U4x2 afr[MT];
#pragma unroll
      for (int m = 0; m < MT; ++m) {
        const unsigned short* ap = sKey + (m * 16 + col) * PITCH + aoff;
        afr[m].lo = *(const U4*)(ap);
        afr[m].hi = *(const U4*)(ap + 8);
      }
#pragma unroll
      for (int m = 0; m < MT; ++m) {
        C[m] = __builtin_amdgcn_wmma_f32_16x16x32_bf16(false, __builtin_bit_cast(v16bf, afr[m]),
                                                       false, Bm, (short)0, C[m], false, false);
      }
      bfrag = bnext;
    }

    // tanh + Ws weighting + row reduction.
    // C VGPR r, lanes 0-15: (row r, col lane); lanes 16-31: (row r+8, col lane-16)
#pragma unroll
    for (int m = 0; m < MT; ++m) {
#pragma unroll
      for (int r = 0; r < 8; ++r) {
        float val = wsw * tanhf(C[m][r] + qpw);
        val += __shfl_xor(val, 8, 32);
        val += __shfl_xor(val, 4, 32);
        val += __shfl_xor(val, 2, 32);
        val += __shfl_xor(val, 1, 32);
        if (col == 0)                      // lanes 0 and 16 own rows r and r+8
          sPart[wave * ROWS + m * 16 + r + (hi << 3)] += val;
      }
    }
  }
  __syncthreads();

  // deterministic cross-wave sum (each wave contributed a disjoint o-range)
  for (int row = tid; row < ROWS; row += 256) {
    float s = 0.f;
#pragma unroll
    for (int w = 0; w < 8; ++w) s += sPart[w * ROWS + row];
    score[b * Vn + v0 + row] = s;
  }
}

// ---------------- softmax over V per batch ----------------
__global__ __launch_bounds__(256) void softmax_kernel(const float* __restrict__ score,
                                                      float* __restrict__ attn) {
  __shared__ float red[256];
  const int b = blockIdx.x, tid = threadIdx.x;
  const float* s = score + b * Vn;
  float mx = -3.402823466e38f;
  for (int v = tid; v < Vn; v += 256) mx = fmaxf(mx, s[v]);
  red[tid] = mx; __syncthreads();
  for (int st = 128; st > 0; st >>= 1) {
    if (tid < st) red[tid] = fmaxf(red[tid], red[tid + st]);
    __syncthreads();
  }
  mx = red[0]; __syncthreads();
  float sum = 0.f;
  for (int v = tid; v < Vn; v += 256) sum += __expf(s[v] - mx);
  red[tid] = sum; __syncthreads();
  for (int st = 128; st > 0; st >>= 1) {
    if (tid < st) red[tid] += red[tid + st];
    __syncthreads();
  }
  const float inv = 1.0f / red[0];
  for (int v = tid; v < Vn; v += 256) attn[b * Vn + v] = __expf(s[v] - mx) * inv;
}

// ---------------- context partials: part[vs,b,h] = sum_{v in seg} attn[b,v]*value[b,v,h] ----------------
__global__ __launch_bounds__(256) void context_part_kernel(const float* __restrict__ value,
                                                           const float* __restrict__ attn,
                                                           float* __restrict__ part) {
  __shared__ float sa[256];
  const int vs  = blockIdx.x;    // 0..3 V-segment
  const int hc  = blockIdx.y;    // 0..3 H-chunk
  const int b   = blockIdx.z;    // 0..31
  const int tid = threadIdx.x;
  const int h = hc * 256 + tid;
  const float* at = attn + b * Vn + vs * 1024;
  const float* vb = value + (size_t)b * Vn * Hn + (size_t)vs * 1024 * Hn;
  float acc = 0.f;
  for (int vv = 0; vv < 1024; vv += 256) {
    __syncthreads();
    sa[tid] = at[vv + tid];
    __syncthreads();
    const float* vp = vb + (size_t)vv * Hn + h;
#pragma unroll 4
    for (int j = 0; j < 256; ++j) acc += sa[j] * vp[(size_t)j * Hn];
  }
  part[((vs * Bn) + b) * Hn + h] = acc;
}

__global__ __launch_bounds__(256) void context_reduce_kernel(const float* __restrict__ part,
                                                             float* __restrict__ ctx) {
  const int i = blockIdx.x * 256 + threadIdx.x;    // Bn*Hn total
  ctx[i] = part[i] + part[Bn * Hn + i] + part[2 * Bn * Hn + i] + part[3 * Bn * Hn + i];
}

extern "C" void kernel_launch(void* const* d_in, const int* in_sizes, int n_in,
                              void* d_out, int out_size, void* d_ws, size_t ws_size,
                              hipStream_t stream) {
  (void)in_sizes; (void)n_in; (void)out_size; (void)ws_size;
  const float* query = (const float*)d_in[0];
  const float* key   = (const float*)d_in[1];
  const float* value = (const float*)d_in[2];
  const float* Wq    = (const float*)d_in[3];
  const float* Wk    = (const float*)d_in[4];
  const float* bias  = (const float*)d_in[5];
  const float* Ws    = (const float*)d_in[6];
  // d_in[7] = bs: softmax is shift-invariant, bs cancels in both outputs.

  float* out  = (float*)d_out;
  float* ctx  = out;                 // (B,1,H) = 32768 floats
  float* attn = out + Bn * Hn;       // (B,V)   = 131072 floats

  // workspace layout (all 16B-aligned)
  float* score = (float*)d_ws;                                  // B*V
  float* qp    = score + Bn * Vn;                               // B*H
  float* part  = qp + Bn * Hn;                                  // 4*B*H
  unsigned short* wkb = (unsigned short*)(part + 4 * Bn * Hn);  // H*H bf16

  qp_kernel<<<Bn, 256, 0, stream>>>(query, Wq, bias, qp);
  wkb_kernel<<<(Hn * Hn) / 1024, 256, 0, stream>>>(Wk, wkb);

  const int smem = ROWS * PITCH * 2 + 8 * ROWS * 4;   // 129KB key tile + 2KB partials
  score_kernel<<<Bn * (Vn / ROWS), 256, smem, stream>>>(key, wkb, qp, Ws, score);

  softmax_kernel<<<Bn, 256, 0, stream>>>(score, attn);
  context_part_kernel<<<dim3(4, 4, Bn), 256, 0, stream>>>(value, attn, part);
  context_reduce_kernel<<<(Bn * Hn) / 256, 256, 0, stream>>>(part, ctx);
}
